// Block_62354335203350
// MI455X (gfx1250) — matrix operand order repaired and checked by reading, not compiled, not run on hardware
//
#include <hip/hip_runtime.h>

// ---------------------------------------------------------------------------
// Transformer block (pre-LN MHA + FFN) for MI455X / gfx1250, wave32 WMMA.
// B=2, T=2048, E=1024, H=16, HS=64.  All matmuls: v_wmma_f32_16x16x32_f16.
// Staging uses GLOBAL_LOAD_ASYNC_TO_LDS_B128 (ASYNCcnt) when available.
// ---------------------------------------------------------------------------

typedef __attribute__((ext_vector_type(16))) _Float16 v16h;
typedef __attribute__((ext_vector_type(4)))  _Float16 v4h;
typedef __attribute__((ext_vector_type(8)))  float    v8f;
typedef __attribute__((ext_vector_type(4)))  int      v4i;

#define Bdim 2
#define Tdim 2048
#define Edim 1024
#define Hdim 16
#define HS   64
#define NTOK (Bdim * Tdim)     // 4096 rows of the residual stream
#define FF   (4 * Edim)        // 4096

// LDS row strides (in halves): 16B-aligned rows, padded vs. 64-bank conflicts
#define GSTR 72   // GEMM tiles (K-chunk = 64 halves + 8 pad)
#define ASTR 72   // attention tiles (64 halves + 8 pad)
#define SSTR 66   // fp32 score tile stride

__device__ __forceinline__ v8f wmma16(v16h a, v16h b, v8f c) {
  // v_wmma_f32_16x16x32_f16  (neg_a, A, neg_b, B, c_mod, C, reuse_a, reuse_b)
  return __builtin_amdgcn_wmma_f32_16x16x32_f16(false, a, false, b, (short)0, c,
                                                false, false);
}

// --- async global->LDS copy (CDNA5), 16B per lane, with sync fallback ------
#if defined(__has_builtin)
#if __has_builtin(__builtin_amdgcn_global_load_async_to_lds_b128)
#define HAS_ASYNC_LDS 1
#endif
#if __has_builtin(__builtin_amdgcn_s_wait_asynccnt)
#define HAS_ASYNC_WAIT 1
#endif
#endif

#define AS1P __attribute__((address_space(1))) v4i*
#define AS3P __attribute__((address_space(3))) v4i*

__device__ __forceinline__ void cp_async16(void* lds, const void* g) {
#ifdef HAS_ASYNC_LDS
  // Param 1: global src as AS(1) int4* (non-const); param 2: LDS dst as
  // AS(3) int4*.  Generic LDS addresses truncate to the AS(3) offset
  // (aperture rule); generic global pointers are value-identical in AS(1).
  __builtin_amdgcn_global_load_async_to_lds_b128(
      (AS1P)(unsigned long long)g,
      (AS3P)(unsigned int)(unsigned long long)lds,
      0, 0);
#else
  *(uint4*)lds = *(const uint4*)g;
#endif
}

__device__ __forceinline__ void async_wait_all() {
#ifdef HAS_ASYNC_WAIT
  __builtin_amdgcn_s_wait_asynccnt(0);
#endif
}

// A fragment, 16x32 f16 (ISA 7.12.2): lane 0-15 row M=lane, halves 0-7 ->
// K 0..7, halves 8-15 -> K 16..23; lanes 16-31 same rows, K +8.
__device__ __forceinline__ v16h load_a_frag(const _Float16* p, int lane,
                                            int stride) {
  union { v16h v; uint4 q[2]; } f;
  const _Float16* r = p + (lane & 15) * stride + ((lane >> 4) * 8);
  f.q[0] = *(const uint4*)(r);
  f.q[1] = *(const uint4*)(r + 16);
  return f.v;
}

// B fragment, 32x16 f16: lanes 0-15 col N=lane hold K 0..7 then K 16..23,
// lanes 16-31 same cols, K +8.  LDS stores B^T ([n][k]) so this is contiguous.
__device__ __forceinline__ v16h load_b_frag(const _Float16* p, int lane,
                                            int stride) {
  union { v16h v; uint4 q[2]; } f;
  const _Float16* r = p + (lane & 15) * stride + ((lane >> 4) * 8);
  f.q[0] = *(const uint4*)(r);
  f.q[1] = *(const uint4*)(r + 16);
  return f.v;
}

// ---------------------------------------------------------------------------
// Weight conversion: f32 -> f16, transposed to [N][K] so B-fragments are
// contiguous in LDS.
// ---------------------------------------------------------------------------
__global__ void convT_kernel(const float* __restrict__ src,
                             _Float16* __restrict__ dst, int K, int N) {
  int idx = blockIdx.x * 256 + threadIdx.x;      // over dst, [N][K]
  if (idx >= K * N) return;
  int n = idx / K;
  int k = idx - n * K;
  dst[idx] = (_Float16)src[(size_t)k * N + n];
}

// Wq/Wk/Wv: [H][E][HS] -> stacked B^T [3*H*HS][E]
__global__ void convQKV_kernel(const float* __restrict__ Wq,
                               const float* __restrict__ Wk,
                               const float* __restrict__ Wv,
                               _Float16* __restrict__ dst) {
  int idx = blockIdx.x * 256 + threadIdx.x;      // 3 * 1024 * 1024 total
  int which = idx >> 20;
  int rem   = idx & ((1 << 20) - 1);
  int n = rem >> 10;                             // head*64 + d
  int e = rem & 1023;
  int head = n >> 6, d = n & 63;
  const float* W = (which == 0) ? Wq : (which == 1) ? Wk : Wv;
  dst[idx] = (_Float16)W[((size_t)head * Edim + e) * HS + d];
}

// ---------------------------------------------------------------------------
// LayerNorm: f32 [NTOK][E] -> f16 [NTOK][E].  One block per row.
// ---------------------------------------------------------------------------
__global__ __launch_bounds__(256) void ln_kernel(const float* __restrict__ x,
                                                 const float* __restrict__ g,
                                                 const float* __restrict__ be,
                                                 _Float16* __restrict__ out) {
  const int row = blockIdx.x;
  const int tid = threadIdx.x;
  const float4 v = ((const float4*)(x + (size_t)row * Edim))[tid];
  float s = v.x + v.y + v.z + v.w;
  float q = v.x * v.x + v.y * v.y + v.z * v.z + v.w * v.w;
  #pragma unroll
  for (int off = 16; off > 0; off >>= 1) {
    s += __shfl_down(s, off);
    q += __shfl_down(q, off);
  }
  __shared__ float rs[8], rq[8];
  __shared__ float mu_s, rstd_s;
  if ((tid & 31) == 0) { rs[tid >> 5] = s; rq[tid >> 5] = q; }
  __syncthreads();
  if (tid == 0) {
    float S = 0.f, Q = 0.f;
    #pragma unroll
    for (int i = 0; i < 8; ++i) { S += rs[i]; Q += rq[i]; }
    float mu = S * (1.0f / Edim);
    mu_s = mu;
    rstd_s = rsqrtf(Q * (1.0f / Edim) - mu * mu + 1e-5f);
  }
  __syncthreads();
  const float mu = mu_s, rstd = rstd_s;
  const float4 gg = ((const float4*)g)[tid];
  const float4 bb = ((const float4*)be)[tid];
  v4h o;
  o[0] = (_Float16)((v.x - mu) * rstd * gg.x + bb.x);
  o[1] = (_Float16)((v.y - mu) * rstd * gg.y + bb.y);
  o[2] = (_Float16)((v.z - mu) * rstd * gg.z + bb.z);
  o[3] = (_Float16)((v.w - mu) * rstd * gg.w + bb.w);
  *(v4h*)(out + (size_t)row * Edim + tid * 4) = o;
}

// ---------------------------------------------------------------------------
// GEMM staging: 128 rows x 64 halves into LDS, 32 halves (4x16B) per thread.
// ---------------------------------------------------------------------------
__device__ __forceinline__ void stage128x64(_Float16* lds,
                                            const _Float16* __restrict__ gsrc,
                                            int ldK, int k0, int tid) {
  const int r = tid >> 1, c = (tid & 1) * 32;
  const _Float16* g = gsrc + (size_t)r * ldK + k0 + c;
  _Float16* d = lds + r * GSTR + c;
  #pragma unroll
  for (int u = 0; u < 4; ++u) cp_async16(d + u * 8, g + u * 8);
}

// ---------------------------------------------------------------------------
// Generic WMMA GEMM:  out[M][N] = A[M][K] * Bt[N][K]^T (+bias)(+relu)(+resid)
// Block tile 128x128, 8 waves (4M x 2N), wave tile 32x64, K-step 64 via LDS.
// ---------------------------------------------------------------------------
template <bool RELU, bool RESID, bool OUT16>
__global__ __launch_bounds__(256) void gemm_kernel(
    const _Float16* __restrict__ A, const _Float16* __restrict__ Bt,
    const float* __restrict__ bias, const float* __restrict__ resid,
    void* __restrict__ out, int M, int N, int K) {
  __shared__ __attribute__((aligned(16))) _Float16 lA[128 * GSTR];
  __shared__ __attribute__((aligned(16))) _Float16 lB[128 * GSTR];
  const int tid = threadIdx.x;
  const int wave = tid >> 5, lane = tid & 31;
  const int m0 = blockIdx.y * 128;
  const int n0 = blockIdx.x * 128;
  const int wm = (wave >> 1) * 32;     // 4 waves in M
  const int wn = (wave & 1) * 64;      // 2 waves in N
  v8f acc[2][4] = {};

  for (int k0 = 0; k0 < K; k0 += 64) {
    stage128x64(lA, A + (size_t)m0 * K, K, k0, tid);
    stage128x64(lB, Bt + (size_t)n0 * K, K, k0, tid);
    async_wait_all();
    __syncthreads();
    #pragma unroll
    for (int kk = 0; kk < 64; kk += 32) {
      v16h a0 = load_a_frag(lA + (wm +  0) * GSTR + kk, lane, GSTR);
      v16h a1 = load_a_frag(lA + (wm + 16) * GSTR + kk, lane, GSTR);
      v16h b0 = load_b_frag(lB + (wn +  0) * GSTR + kk, lane, GSTR);
      v16h b1 = load_b_frag(lB + (wn + 16) * GSTR + kk, lane, GSTR);
      v16h b2 = load_b_frag(lB + (wn + 32) * GSTR + kk, lane, GSTR);
      v16h b3 = load_b_frag(lB + (wn + 48) * GSTR + kk, lane, GSTR);
      acc[0][0] = wmma16(a0, b0, acc[0][0]);
      acc[0][1] = wmma16(a0, b1, acc[0][1]);
      acc[0][2] = wmma16(a0, b2, acc[0][2]);
      acc[0][3] = wmma16(a0, b3, acc[0][3]);
      acc[1][0] = wmma16(a1, b0, acc[1][0]);
      acc[1][1] = wmma16(a1, b1, acc[1][1]);
      acc[1][2] = wmma16(a1, b2, acc[1][2]);
      acc[1][3] = wmma16(a1, b3, acc[1][3]);
    }
    __syncthreads();
  }
  // Epilogue.  C layout: VGPR r -> row = r + 8*(lane>>4), col = lane&15.
  #pragma unroll
  for (int i = 0; i < 2; ++i) {
    #pragma unroll
    for (int j = 0; j < 4; ++j) {
      const int col = n0 + wn + j * 16 + (lane & 15);
      const int rb  = m0 + wm + i * 16 + 8 * (lane >> 4);
      const float bv = bias ? bias[col] : 0.0f;
      #pragma unroll
      for (int r = 0; r < 8; ++r) {
        const int row = rb + r;
        float v = acc[i][j][r] + bv;
        if (RELU)  v = fmaxf(v, 0.0f);
        if (RESID) v += resid[(size_t)row * N + col];
        if (OUT16) ((_Float16*)out)[(size_t)row * N + col] = (_Float16)v;
        else       ((float*)out)[(size_t)row * N + col] = v;
      }
    }
  }
}

// ---------------------------------------------------------------------------
// Fused QKV GEMM: A = ln(x) f16 [NTOK][E], Bt = [3072][E].
// Scatter: Q,K row-major [B*H][T][HS]; V transposed [B*H][HS][T].
// ---------------------------------------------------------------------------
__global__ __launch_bounds__(256) void gemm_qkv_kernel(
    const _Float16* __restrict__ A, const _Float16* __restrict__ Bt,
    _Float16* __restrict__ Qo, _Float16* __restrict__ Ko,
    _Float16* __restrict__ Vto) {
  __shared__ __attribute__((aligned(16))) _Float16 lA[128 * GSTR];
  __shared__ __attribute__((aligned(16))) _Float16 lB[128 * GSTR];
  const int tid = threadIdx.x;
  const int wave = tid >> 5, lane = tid & 31;
  const int m0 = blockIdx.y * 128;
  const int n0 = blockIdx.x * 128;
  const int wm = (wave >> 1) * 32;
  const int wn = (wave & 1) * 64;
  const int K = Edim;
  v8f acc[2][4] = {};

  for (int k0 = 0; k0 < K; k0 += 64) {
    stage128x64(lA, A + (size_t)m0 * K, K, k0, tid);
    stage128x64(lB, Bt + (size_t)n0 * K, K, k0, tid);
    async_wait_all();
    __syncthreads();
    #pragma unroll
    for (int kk = 0; kk < 64; kk += 32) {
      v16h a0 = load_a_frag(lA + (wm +  0) * GSTR + kk, lane, GSTR);
      v16h a1 = load_a_frag(lA + (wm + 16) * GSTR + kk, lane, GSTR);
      #pragma unroll
      for (int j = 0; j < 4; ++j) {
        v16h bj = load_b_frag(lB + (wn + j * 16) * GSTR + kk, lane, GSTR);
        acc[0][j] = wmma16(a0, bj, acc[0][j]);
        acc[1][j] = wmma16(a1, bj, acc[1][j]);
      }
    }
    __syncthreads();
  }
  #pragma unroll
  for (int i = 0; i < 2; ++i) {
    #pragma unroll
    for (int j = 0; j < 4; ++j) {
      const int colg = n0 + wn + j * 16 + (lane & 15);   // 0..3071
      const int which = colg >> 10;
      const int within = colg & 1023;
      const int head = within >> 6, d = within & 63;
      #pragma unroll
      for (int r = 0; r < 8; ++r) {
        const int rowg = m0 + wm + i * 16 + 8 * (lane >> 4) + r;
        const int b = rowg >> 11, t = rowg & (Tdim - 1);
        const int bh = b * Hdim + head;
        const _Float16 v = (_Float16)acc[i][j][r];
        if (which == 0)      Qo[((size_t)bh * Tdim + t) * HS + d] = v;
        else if (which == 1) Ko[((size_t)bh * Tdim + t) * HS + d] = v;
        else                 Vto[((size_t)bh * HS + d) * Tdim + t] = v;
      }
    }
  }
}

// ---------------------------------------------------------------------------
// Flash attention (causal), 64-query tile per block, online softmax.
// S waves: 2M x 4N over 64x64.  O waves: 4M x 2N (16x32 per wave).
// ---------------------------------------------------------------------------
__global__ __launch_bounds__(256) void attn_kernel(
    const _Float16* __restrict__ Qb,   // [B*H][T][HS]
    const _Float16* __restrict__ Kb,   // [B*H][T][HS]
    const _Float16* __restrict__ Vt,   // [B*H][HS][T]
    _Float16* __restrict__ Ob) {       // [B][T][H*HS]
  __shared__ __attribute__((aligned(16))) _Float16 sQ[64 * ASTR];
  __shared__ __attribute__((aligned(16))) _Float16 sK[64 * ASTR];
  __shared__ __attribute__((aligned(16))) _Float16 sV[64 * ASTR];
  __shared__ __attribute__((aligned(16))) _Float16 sP[64 * ASTR];
  __shared__ float sS[64 * SSTR];
  __shared__ float sM[64], sL[64], sAl[64];

  const int tid = threadIdx.x;
  const int wave = tid >> 5, lane = tid & 31;
  const int bh = blockIdx.y;
  const int b = bh >> 4, h = bh & 15;
  const int tq0 = blockIdx.x * 64;

  {  // Q tile resident for the whole block (64x64 halves, 16/thread)
    const int r = tid >> 2, cg = (tid & 3) * 16;
    const _Float16* g = Qb + ((size_t)bh * Tdim + tq0 + r) * HS + cg;
    _Float16* d = sQ + r * ASTR + cg;
    cp_async16(d, g);
    cp_async16(d + 8, g + 8);
  }
  if (tid < 64) { sM[tid] = -INFINITY; sL[tid] = 0.0f; }

  const int srow0 = (wave >> 2) * 32, scol0 = (wave & 3) * 16;
  const int orow0 = (wave >> 1) * 16, ocol0 = (wave & 1) * 32;
  v8f o[2] = {};
  async_wait_all();
  __syncthreads();

  for (int s0 = 0; s0 <= tq0; s0 += 64) {
    {  // stage K (row-major) and V^T tiles
      const int r = tid >> 2, cg = (tid & 3) * 16;
      const _Float16* kg = Kb + ((size_t)bh * Tdim + s0 + r) * HS + cg;
      _Float16* kd = sK + r * ASTR + cg;
      cp_async16(kd, kg);
      cp_async16(kd + 8, kg + 8);
      const _Float16* vg = Vt + ((size_t)bh * HS + r) * Tdim + s0 + cg;
      _Float16* vd = sV + r * ASTR + cg;
      cp_async16(vd, vg);
      cp_async16(vd + 8, vg + 8);
    }
    async_wait_all();
    __syncthreads();

    // S = (Q K^T) * HS^-0.5, causal mask on diagonal tile
    v8f sacc[2] = {};
    #pragma unroll
    for (int kk = 0; kk < HS; kk += 32) {
      v16h bf = load_b_frag(sK + scol0 * ASTR + kk, lane, ASTR);
      v16h a0 = load_a_frag(sQ + (srow0 +  0) * ASTR + kk, lane, ASTR);
      v16h a1 = load_a_frag(sQ + (srow0 + 16) * ASTR + kk, lane, ASTR);
      sacc[0] = wmma16(a0, bf, sacc[0]);
      sacc[1] = wmma16(a1, bf, sacc[1]);
    }
    const bool diag = (s0 == tq0);
    #pragma unroll
    for (int i = 0; i < 2; ++i) {
      const int rb = srow0 + i * 16 + 8 * (lane >> 4);
      const int col = scol0 + (lane & 15);
      #pragma unroll
      for (int r = 0; r < 8; ++r) {
        float v = sacc[i][r] * 0.125f;           // 1/sqrt(64)
        if (diag && col > rb + r) v = -INFINITY;
        sS[(rb + r) * SSTR + col] = v;
      }
    }
    __syncthreads();

    {  // online softmax: 4 lanes per row
      const int row = tid >> 2, seg = (tid & 3) * 16;
      float vals[16];
      float mloc = -INFINITY;
      #pragma unroll
      for (int j = 0; j < 16; ++j) {
        vals[j] = sS[row * SSTR + seg + j];
        mloc = fmaxf(mloc, vals[j]);
      }
      mloc = fmaxf(mloc, __shfl_xor(mloc, 1));
      mloc = fmaxf(mloc, __shfl_xor(mloc, 2));
      const float mold = sM[row];
      const float mnew = fmaxf(mold, mloc);
      float ssum = 0.0f;
      #pragma unroll
      for (int j = 0; j < 16; ++j) {
        const float p = __expf(vals[j] - mnew);
        ssum += p;
        sP[row * ASTR + seg + j] = (_Float16)p;
      }
      ssum += __shfl_xor(ssum, 1);
      ssum += __shfl_xor(ssum, 2);
      if ((tid & 3) == 0) {
        const float al = __expf(mold - mnew);
        sAl[row] = al;
        sL[row] = sL[row] * al + ssum;
        sM[row] = mnew;
      }
    }
    __syncthreads();

    {  // rescale O, then O += P * V
      const int rb = orow0 + 8 * (lane >> 4);
      #pragma unroll
      for (int r = 0; r < 8; ++r) {
        const float al = sAl[rb + r];
        o[0][r] *= al;
        o[1][r] *= al;
      }
    }
    #pragma unroll
    for (int kk = 0; kk < 64; kk += 32) {
      v16h pa = load_a_frag(sP + orow0 * ASTR + kk, lane, ASTR);
      v16h v0 = load_b_frag(sV + (ocol0 +  0) * ASTR + kk, lane, ASTR);
      v16h v1 = load_b_frag(sV + (ocol0 + 16) * ASTR + kk, lane, ASTR);
      o[0] = wmma16(pa, v0, o[0]);
      o[1] = wmma16(pa, v1, o[1]);
    }
    __syncthreads();
  }

  #pragma unroll
  for (int j = 0; j < 2; ++j) {
    const int col = ocol0 + j * 16 + (lane & 15);
    #pragma unroll
    for (int r = 0; r < 8; ++r) {
      const int row = orow0 + 8 * (lane >> 4) + r;
      const float v = o[j][r] / sL[row];
      Ob[((size_t)b * Tdim + tq0 + row) * Edim + h * HS + col] = (_Float16)v;
    }
  }
}

// ---------------------------------------------------------------------------
// Host-side orchestration
// ---------------------------------------------------------------------------
extern "C" void kernel_launch(void* const* d_in, const int* in_sizes, int n_in,
                              void* d_out, int out_size, void* d_ws,
                              size_t ws_size, hipStream_t stream) {
  const float* x     = (const float*)d_in[0];
  const float* Wq    = (const float*)d_in[1];
  const float* Wk    = (const float*)d_in[2];
  const float* Wv    = (const float*)d_in[3];
  const float* Wp    = (const float*)d_in[4];
  const float* bp    = (const float*)d_in[5];
  const float* W1    = (const float*)d_in[6];
  const float* b1    = (const float*)d_in[7];
  const float* W2    = (const float*)d_in[8];
  const float* b2    = (const float*)d_in[9];
  const float* g1    = (const float*)d_in[10];
  const float* beta1 = (const float*)d_in[11];
  const float* g2    = (const float*)d_in[12];
  const float* beta2 = (const float*)d_in[13];
  float* out = (float*)d_out;

  // Workspace carve-up (all sizes 256B-multiples)
  char* p = (char*)d_ws;
  auto take = [&](size_t bytes) { char* r = p; p += bytes; return r; };
  _Float16* hf16   = (_Float16*)take((size_t)NTOK * Edim * 2);        // ln1(x)
  _Float16* WqkvT  = (_Float16*)take((size_t)3 * Edim * Edim * 2);
  _Float16* WpT    = (_Float16*)take((size_t)Edim * Edim * 2);
  _Float16* W1T    = (_Float16*)take((size_t)Edim * FF * 2);
  _Float16* W2T    = (_Float16*)take((size_t)FF * Edim * 2);
  _Float16* Qb     = (_Float16*)take((size_t)NTOK * Edim * 2);
  _Float16* Kb     = (_Float16*)take((size_t)NTOK * Edim * 2);
  _Float16* Vtb    = (_Float16*)take((size_t)NTOK * Edim * 2);
  _Float16* attn_o = (_Float16*)take((size_t)NTOK * Edim * 2);
  float*    x1     = (float*)take((size_t)NTOK * Edim * 4);           // resid 1
  _Float16* h2     = (_Float16*)take((size_t)NTOK * Edim * 2);        // ln2(x1)
  _Float16* ffmid  = (_Float16*)take((size_t)NTOK * FF * 2);
  (void)ws_size; (void)in_sizes; (void)n_in; (void)out_size;

  // 1) weight conversion to f16 transposed layouts
  convQKV_kernel<<<(3 * Edim * Edim) / 256, 256, 0, stream>>>(Wq, Wk, Wv, WqkvT);
  convT_kernel<<<(Edim * Edim) / 256, 256, 0, stream>>>(Wp, WpT, Edim, Edim);
  convT_kernel<<<(Edim * FF) / 256, 256, 0, stream>>>(W1, W1T, Edim, FF);
  convT_kernel<<<(FF * Edim) / 256, 256, 0, stream>>>(W2, W2T, FF, Edim);

  // 2) LN1
  ln_kernel<<<NTOK, 256, 0, stream>>>(x, g1, beta1, hf16);

  // 3) fused QKV projection (N = 3072), scatter Q/K row-major, V transposed
  gemm_qkv_kernel<<<dim3(3 * Edim / 128, NTOK / 128), 256, 0, stream>>>(
      hf16, WqkvT, Qb, Kb, Vtb);

  // 4) causal flash attention
  attn_kernel<<<dim3(Tdim / 64, Bdim * Hdim), 256, 0, stream>>>(Qb, Kb, Vtb,
                                                                attn_o);

  // 5) output projection + bias + residual(x) -> x1 (f32)
  gemm_kernel<false, true, false>
      <<<dim3(Edim / 128, NTOK / 128), 256, 0, stream>>>(
          attn_o, WpT, bp, x, (void*)x1, NTOK, Edim, Edim);

  // 6) LN2
  ln_kernel<<<NTOK, 256, 0, stream>>>(x1, g2, beta2, h2);

  // 7) FFN up-projection + bias + ReLU -> f16
  gemm_kernel<true, false, true>
      <<<dim3(FF / 128, NTOK / 128), 256, 0, stream>>>(
          h2, W1T, b1, nullptr, (void*)ffmid, NTOK, FF, Edim);

  // 8) FFN down-projection + bias + residual(x1) -> d_out (f32)
  gemm_kernel<false, true, false>
      <<<dim3(Edim / 128, NTOK / 128), 256, 0, stream>>>(
          ffmid, W2T, b2, x1, (void*)out, NTOK, Edim, FF);
}
